// BohaoDecoder_1529008357493
// MI455X (gfx1250) — matrix operand order repaired and checked by reading, not compiled, not run on hardware
//
#include <hip/hip_runtime.h>
#include <hip/hip_bf16.h>
#include <stdint.h>

// ---------------------------------------------------------------------------
// Tacotron2 decoder for MI455X (gfx1250): bf16 WMMA for all large GEMMs,
// per-step launch chain (graph-captured by harness).
// All GEMM K dims are multiples of 32 (prenet K=80 zero-padded to 96), so the
// WMMA K-loop is guard-free: 4x global_load_b128 + v_wmma per 32-K step.
// ---------------------------------------------------------------------------

typedef __bf16 bf16;
typedef __attribute__((ext_vector_type(16))) __bf16 v16bf;
typedef __attribute__((ext_vector_type(8)))  float  v8f;

#define B_       64
#define T_IN_    512
#define T_OUT_   500
#define N_MEL_   80
#define PREK_    96      // N_MEL_ zero-padded to a multiple of 32
#define ENC_     512
#define ATT_DIM_ 128
#define RNN_     1024
#define PRE_     256
#define GATES_   4096

union Frag16 {
    v16bf v;
    uint4 u4[2];
};

// ---------------------------------------------------------------------------
// Generic WMMA GEMM:  C[M,N] (fp32) = A[M,K](bf16,row-major,lda) *
//                     W[N, colofs..colofs+K](bf16,row-major,ldw)^T  (+bias / +=C)
// One wave per 16x16 C tile; K-loop in steps of 32 with v_wmma_f32_16x16x32_bf16.
// grid.x covers N/16 tiles in groups of 8 waves, grid.y covers M/16 tiles.
// REQUIRES: M,N multiples of 16; K multiple of 32; A,W 16B-aligned rows
// (lda, ldw, colofs all multiples of 8 elements).
// ---------------------------------------------------------------------------
__global__ __launch_bounds__(256)
void wmma_gemm_bf16(const bf16* __restrict__ A, int lda,
                    const bf16* __restrict__ W, int ldw, int colofs,
                    const float* __restrict__ bias,
                    float* __restrict__ C, int ldc,
                    int M, int N, int K, int accumulate)
{
    const int lane  = threadIdx.x & 31;
    const int wave  = threadIdx.x >> 5;
    const int ntile = blockIdx.x * 8 + wave;
    const int mtile = blockIdx.y;
    const int n0 = ntile * 16;
    const int m0 = mtile * 16;
    if (n0 >= N || m0 >= M) return;          // wave-uniform: EXEC stays all-ones

    const int half = lane >> 4;
    const int l15  = lane & 15;

    // Per-lane base pointers for the two 16B runs of each fragment.
    // A frag (16x32): VGPR r<4 hold K = 2r + 8*half ; r>=4 hold K = 16 + 2(r-4) + 8*half
    //   -> contiguous runs at k0 + 8*half and k0 + 16 + 8*half (8 bf16 each).
    // B frag (32x16): same K order as A
    //   -> contiguous runs at k0 + 8*half and k0 + 16 + 8*half.
    const bf16* pa = A + (long long)(m0 + l15) * lda + 8 * half;
    const bf16* pb = W + (long long)(n0 + l15) * ldw + colofs + 8 * half;

    v8f acc;
    if (accumulate) {
#pragma unroll
        for (int r = 0; r < 8; ++r)
            acc[r] = C[(long long)(m0 + r + 8 * half) * ldc + n0 + l15];
    } else {
        float b0 = bias ? bias[n0 + l15] : 0.0f;
#pragma unroll
        for (int r = 0; r < 8; ++r) acc[r] = b0;
    }

    for (int k0 = 0; k0 < K; k0 += 32) {
        Frag16 a, b;
        a.u4[0] = *(const uint4*)(pa + k0);
        a.u4[1] = *(const uint4*)(pa + k0 + 16);
        b.u4[0] = *(const uint4*)(pb + k0);
        b.u4[1] = *(const uint4*)(pb + k0 + 16);
        acc = __builtin_amdgcn_wmma_f32_16x16x32_bf16(
            false, a.v, false, b.v, (short)0, acc, false, false);
    }

#pragma unroll
    for (int r = 0; r < 8; ++r)
        C[(long long)(m0 + r + 8 * half) * ldc + n0 + l15] = acc[r];
}

// ---------------------------------------------------------------------------
// Elementwise helpers
// ---------------------------------------------------------------------------
__global__ void cvt_f32_bf16(const float* __restrict__ in, bf16* __restrict__ out, long long n)
{
    long long i = (long long)blockIdx.x * blockDim.x + threadIdx.x;
    if (i < n) out[i] = (bf16)in[i];
}

// fp32 [rows, incols] -> bf16 [rows, outcols] zero-padded (outcols >= incols)
__global__ void cvt_pad_bf16(const float* __restrict__ in, bf16* __restrict__ out,
                             int rows, int incols, int outcols)
{
    long long i = (long long)blockIdx.x * blockDim.x + threadIdx.x;
    if (i >= (long long)rows * outcols) return;
    int r = (int)(i / outcols), c = (int)(i % outcols);
    out[i] = (bf16)((c < incols) ? in[(long long)r * incols + c] : 0.0f);
}

__global__ void zero_u32(unsigned* __restrict__ p, long long n)
{
    long long i = (long long)blockIdx.x * blockDim.x + threadIdx.x;
    if (i < n) p[i] = 0u;
}

// decoder_inputs [B, N_MEL, T_OUT] -> di rows [(t*B+b), 0..95] (zero-padded,
// go-frame at t==0)
__global__ void build_di(const float* __restrict__ dec_in, bf16* __restrict__ di)
{
    long long i = (long long)blockIdx.x * blockDim.x + threadIdx.x;  // t*B*96 + b*96 + m
    if (i >= (long long)T_OUT_ * B_ * PREK_) return;
    int t = (int)(i / (B_ * PREK_));
    int r = (int)(i % (B_ * PREK_));
    int b = r / PREK_, m = r % PREK_;
    float v = (t == 0 || m >= N_MEL_) ? 0.0f
              : dec_in[((long long)b * N_MEL_ + m) * T_OUT_ + (t - 1)];
    di[i] = (bf16)v;
}

// relu + deterministic 50% "dropout" (*2) -> bf16
__global__ void relu_drop_bf16(const float* __restrict__ in, bf16* __restrict__ out,
                               long long n, unsigned seed)
{
    long long i = (long long)blockIdx.x * blockDim.x + threadIdx.x;
    if (i >= n) return;
    float v = fmaxf(in[i], 0.0f);
    unsigned h = ((unsigned)i * 2654435761u) ^ seed;
    h ^= h >> 13; h *= 0x5bd1e995u; h ^= h >> 15;
    out[i] = (bf16)((h & 1u) ? v * 2.0f : 0.0f);
}

// LSTM cell: gates[B,4096] fp32 -> h,c (fp32) + h bf16.  bhh added here (bih in GEMM bias).
__global__ void lstm_cell(const float* __restrict__ gates, const float* __restrict__ bhh,
                          float* __restrict__ h, float* __restrict__ c,
                          bf16* __restrict__ h_bf)
{
    int i = blockIdx.x * blockDim.x + threadIdx.x;       // B*1024
    if (i >= B_ * RNN_) return;
    int b = i >> 10, j = i & (RNN_ - 1);
    const float* g = gates + (long long)b * GATES_;
    float gi = g[j]            + bhh[j];
    float gf = g[j + RNN_]     + bhh[j + RNN_];
    float gg = g[j + 2 * RNN_] + bhh[j + 2 * RNN_];
    float go = g[j + 3 * RNN_] + bhh[j + 3 * RNN_];
    float si = 1.0f / (1.0f + __expf(-gi));
    float sf = 1.0f / (1.0f + __expf(-gf));
    float so = 1.0f / (1.0f + __expf(-go));
    float cn = sf * c[i] + si * tanhf(gg);
    float hn = so * tanhf(cn);
    c[i] = cn; h[i] = hn; h_bf[i] = (bf16)hn;
}

// ---------------------------------------------------------------------------
// Location-sensitive attention, one workgroup (256 thr) per batch element.
// conv(aw,awc) -> energies = v . tanh(pq + loc@loc_dense^T + pm) -> masked
// softmax -> context; writes aw/awc/ctx(+bf16) and the alignment output row.
// ---------------------------------------------------------------------------
__global__ __launch_bounds__(256)
void attention_step(const float* __restrict__ ah, const float* __restrict__ wq,
                    const float* __restrict__ loc_conv, const float* __restrict__ loc_dense,
                    const float* __restrict__ vvec, const float* __restrict__ pm,
                    const float* __restrict__ memory, const int* __restrict__ mlen,
                    float* __restrict__ aw, float* __restrict__ awc,
                    float* __restrict__ ctx, bf16* __restrict__ ctx_bf,
                    float* __restrict__ align_out, int t)
{
    __shared__ float s_aw[T_IN_], s_awc[T_IN_], s_pq[ATT_DIM_], s_e[T_IN_], s_red[256];
    const int b = blockIdx.x, tid = threadIdx.x;

    for (int i = tid; i < T_IN_; i += 256) { s_aw[i] = aw[b * T_IN_ + i]; s_awc[i] = awc[b * T_IN_ + i]; }
    __syncthreads();

    if (tid < ATT_DIM_) {                       // pq = ah @ wq^T   (128 x 1024 dot)
        const float* hr = ah + (long long)b * RNN_;
        const float* wr = wq + (long long)tid * RNN_;
        float s = 0.0f;
        for (int j = 0; j < RNN_; ++j) s += hr[j] * wr[j];
        s_pq[tid] = s;
    }
    __syncthreads();

    const int len = mlen[b];
    for (int t0 = tid; t0 < T_IN_; t0 += 256) {
        float loc[32];
#pragma unroll
        for (int f = 0; f < 32; ++f) loc[f] = 0.0f;
        for (int k = 0; k < 31; ++k) {          // conv, pad 15
            int p = t0 + k - 15;
            float a0 = (p >= 0 && p < T_IN_) ? s_aw[p]  : 0.0f;
            float a1 = (p >= 0 && p < T_IN_) ? s_awc[p] : 0.0f;
#pragma unroll
            for (int f = 0; f < 32; ++f)
                loc[f] += loc_conv[(f * 2 + 0) * 31 + k] * a0
                        + loc_conv[(f * 2 + 1) * 31 + k] * a1;
        }
        const float* pmr = pm + ((long long)b * T_IN_ + t0) * ATT_DIM_;
        float e = 0.0f;
        for (int a = 0; a < ATT_DIM_; ++a) {
            float s = s_pq[a] + pmr[a];
            const float* ld = loc_dense + a * 32;
#pragma unroll
            for (int f = 0; f < 32; ++f) s += ld[f] * loc[f];
            e += vvec[a] * tanhf(s);
        }
        s_e[t0] = (t0 < len) ? e : -1e30f;
    }
    __syncthreads();

    // softmax over 512 entries
    s_red[tid] = fmaxf(s_e[tid], s_e[tid + 256]);
    __syncthreads();
    for (int s = 128; s > 0; s >>= 1) { if (tid < s) s_red[tid] = fmaxf(s_red[tid], s_red[tid + s]); __syncthreads(); }
    float mx = s_red[0];
    __syncthreads();
    s_e[tid]       = __expf(s_e[tid]       - mx);
    s_e[tid + 256] = __expf(s_e[tid + 256] - mx);
    __syncthreads();
    s_red[tid] = s_e[tid] + s_e[tid + 256];
    __syncthreads();
    for (int s = 128; s > 0; s >>= 1) { if (tid < s) s_red[tid] += s_red[tid + s]; __syncthreads(); }
    float inv = 1.0f / s_red[0];
    __syncthreads();

    for (int i = tid; i < T_IN_; i += 256) {
        float w  = s_e[i] * inv;
        float wc = s_awc[i] + w;
        s_aw[i] = w;
        aw[b * T_IN_ + i]  = w;
        awc[b * T_IN_ + i] = wc;
        align_out[((long long)b * T_OUT_ + t) * T_IN_ + i] = w;
    }
    __syncthreads();

    for (int d = tid; d < ENC_; d += 256) {     // ctx = aw @ memory[b]
        const float* mr = memory + ((long long)b * T_IN_) * ENC_ + d;
        float s = 0.0f;
        for (int tt = 0; tt < T_IN_; ++tt) s += s_aw[tt] * mr[(long long)tt * ENC_];
        ctx[b * ENC_ + d] = s;
        ctx_bf[b * ENC_ + d] = (bf16)s;
    }
}

// projection + gate: dhc = [dh(1024), ctx(512)] -> mel[80] & gate scalar
__global__ void proj_step(const float* __restrict__ dh, const float* __restrict__ ctx,
                          const float* __restrict__ proj_w, const float* __restrict__ proj_b,
                          const float* __restrict__ gate_w, const float* __restrict__ gate_b,
                          float* __restrict__ out, int t)
{
    const int b = blockIdx.x, o = threadIdx.x;
    if (o > 80) return;
    const float* w = (o < 80) ? (proj_w + (long long)o * (RNN_ + ENC_)) : gate_w;
    float s = (o < 80) ? proj_b[o] : gate_b[0];
    const float* hr = dh  + (long long)b * RNN_;
    const float* cr = ctx + (long long)b * ENC_;
    for (int j = 0; j < RNN_; ++j) s += hr[j] * w[j];
    for (int d = 0; d < ENC_; ++d) s += cr[d] * w[RNN_ + d];
    if (o < 80) out[((long long)b * T_OUT_ + t) * N_MEL_ + o] = s;
    else        out[(long long)B_ * T_OUT_ * N_MEL_ + (long long)b * T_OUT_ + t] = s;
}

// ---------------------------------------------------------------------------
// Host driver
// ---------------------------------------------------------------------------
static inline size_t alignup(size_t x) { return (x + 255) & ~(size_t)255; }

extern "C" void kernel_launch(void* const* d_in, const int* in_sizes, int n_in,
                              void* d_out, int out_size, void* d_ws, size_t ws_size,
                              hipStream_t stream)
{
    const float* memory   = (const float*)d_in[0];
    const float* dec_in   = (const float*)d_in[1];
    const int*   mlen     = (const int*)  d_in[2];
    const float* pre_w1   = (const float*)d_in[3];
    const float* pre_w2   = (const float*)d_in[4];
    const float* att_wih  = (const float*)d_in[5];
    const float* att_whh  = (const float*)d_in[6];
    const float* att_bih  = (const float*)d_in[7];
    const float* att_bhh  = (const float*)d_in[8];
    const float* wq       = (const float*)d_in[9];
    const float* wm       = (const float*)d_in[10];
    const float* vvec     = (const float*)d_in[11];
    const float* loc_conv = (const float*)d_in[12];
    const float* loc_dense= (const float*)d_in[13];
    const float* dec_wih  = (const float*)d_in[14];
    const float* dec_whh  = (const float*)d_in[15];
    const float* dec_bih  = (const float*)d_in[16];
    const float* dec_bhh  = (const float*)d_in[17];
    const float* proj_w   = (const float*)d_in[18];
    const float* proj_b   = (const float*)d_in[19];
    const float* gate_w   = (const float*)d_in[20];
    const float* gate_b   = (const float*)d_in[21];
    float* out = (float*)d_out;

    // ---- workspace layout ----
    char* ws = (char*)d_ws;
    size_t off = 0;
    auto take = [&](size_t bytes) { size_t o = off; off += alignup(bytes); return o; };

    size_t o_att_wih = take((size_t)GATES_ * 768  * 2);
    size_t o_att_whh = take((size_t)GATES_ * RNN_ * 2);
    size_t o_dec_wih = take((size_t)GATES_ * 1536 * 2);
    size_t o_dec_whh = take((size_t)GATES_ * RNN_ * 2);
    size_t o_pre1    = take((size_t)PRE_ * PREK_  * 2);   // zero-padded K
    size_t o_pre2    = take((size_t)PRE_ * PRE_   * 2);
    size_t o_wm      = take((size_t)ATT_DIM_ * ENC_ * 2);
    size_t o_pm      = take((size_t)B_ * T_IN_ * ATT_DIM_ * 4);
    size_t o_x       = take((size_t)T_OUT_ * B_ * PRE_ * 2);
    size_t o_gates   = take((size_t)B_ * GATES_ * 4);
    // fp32 state block (zeroed each call): ah, ac, dh, dc, ctx, aw, awc
    size_t o_state   = take((size_t)(4 * B_ * RNN_ + 3 * B_ * T_IN_) * 4);
    size_t o_ah = o_state;
    size_t o_ac = o_ah + (size_t)B_ * RNN_ * 4;
    size_t o_dh = o_ac + (size_t)B_ * RNN_ * 4;
    size_t o_dc = o_dh + (size_t)B_ * RNN_ * 4;
    size_t o_ctx = o_dc + (size_t)B_ * RNN_ * 4;
    size_t o_aw  = o_ctx + (size_t)B_ * ENC_ * 4;
    size_t o_awc = o_aw + (size_t)B_ * T_IN_ * 4;
    // bf16 state block (zeroed each call): ah_bf, dh_bf, ctx_bf
    size_t o_sbf   = take((size_t)(2 * B_ * RNN_ + B_ * ENC_) * 2);
    size_t o_ah_bf  = o_sbf;
    size_t o_dh_bf  = o_ah_bf + (size_t)B_ * RNN_ * 2;
    size_t o_ctx_bf = o_dh_bf + (size_t)B_ * RNN_ * 2;
    // setup-only scratch (memory_bf16 aliases di+pre_tmp; used at different times)
    size_t sc_mem = (size_t)B_ * T_IN_ * ENC_ * 2;                       // 33.5 MB
    size_t sc_pre = alignup((size_t)T_OUT_ * B_ * PREK_ * 2)             // di bf16 (padded)
                  + alignup((size_t)T_OUT_ * B_ * PRE_ * 4);             // pre_tmp fp32
    size_t o_scratch = take(sc_mem > sc_pre ? sc_mem : sc_pre);
    size_t o_di   = o_scratch;
    size_t o_ptmp = o_scratch + alignup((size_t)T_OUT_ * B_ * PREK_ * 2);
    if (off > ws_size) return;   // workspace too small: do nothing (deterministic)

    bf16*  mem_bf  = (bf16*)(ws + o_scratch);
    bf16*  wih_a   = (bf16*)(ws + o_att_wih);
    bf16*  whh_a   = (bf16*)(ws + o_att_whh);
    bf16*  wih_d   = (bf16*)(ws + o_dec_wih);
    bf16*  whh_d   = (bf16*)(ws + o_dec_whh);
    bf16*  p1_bf   = (bf16*)(ws + o_pre1);
    bf16*  p2_bf   = (bf16*)(ws + o_pre2);
    bf16*  wm_bf   = (bf16*)(ws + o_wm);
    float* pm      = (float*)(ws + o_pm);
    bf16*  x_bf    = (bf16*)(ws + o_x);
    float* gates   = (float*)(ws + o_gates);
    float* ah  = (float*)(ws + o_ah);  float* ac = (float*)(ws + o_ac);
    float* dh  = (float*)(ws + o_dh);  float* dc = (float*)(ws + o_dc);
    float* ctx = (float*)(ws + o_ctx);
    float* aw  = (float*)(ws + o_aw);  float* awc = (float*)(ws + o_awc);
    bf16* ah_bf = (bf16*)(ws + o_ah_bf);
    bf16* dh_bf = (bf16*)(ws + o_dh_bf);
    bf16* ctx_bf = (bf16*)(ws + o_ctx_bf);
    bf16*  di   = (bf16*)(ws + o_di);
    float* ptmp = (float*)(ws + o_ptmp);

    auto grid1 = [](long long n) { return dim3((unsigned)((n + 255) / 256)); };

    // ---- setup: bf16 conversions ----
    long long n;
    n = (long long)B_ * T_IN_ * ENC_;   cvt_f32_bf16<<<grid1(n), 256, 0, stream>>>(memory,  mem_bf, n);
    n = (long long)ATT_DIM_ * ENC_;     cvt_f32_bf16<<<grid1(n), 256, 0, stream>>>(wm,      wm_bf,  n);
    n = (long long)GATES_ * 768;        cvt_f32_bf16<<<grid1(n), 256, 0, stream>>>(att_wih, wih_a,  n);
    n = (long long)GATES_ * RNN_;       cvt_f32_bf16<<<grid1(n), 256, 0, stream>>>(att_whh, whh_a,  n);
    n = (long long)GATES_ * 1536;       cvt_f32_bf16<<<grid1(n), 256, 0, stream>>>(dec_wih, wih_d,  n);
    n = (long long)GATES_ * RNN_;       cvt_f32_bf16<<<grid1(n), 256, 0, stream>>>(dec_whh, whh_d,  n);
    n = (long long)PRE_ * PREK_;        cvt_pad_bf16<<<grid1(n), 256, 0, stream>>>(pre_w1, p1_bf, PRE_, N_MEL_, PREK_);
    n = (long long)PRE_ * PRE_;         cvt_f32_bf16<<<grid1(n), 256, 0, stream>>>(pre_w2,  p2_bf,  n);

    // processed_memory = memory @ wm^T : M=32768, N=128, K=512
    wmma_gemm_bf16<<<dim3(1, (B_ * T_IN_) / 16), 256, 0, stream>>>(
        mem_bf, ENC_, wm_bf, ENC_, 0, nullptr, pm, ATT_DIM_,
        B_ * T_IN_, ATT_DIM_, ENC_, 0);

    // prenet over all 500 rows (go-frame at t=0), K padded 80->96
    n = (long long)T_OUT_ * B_ * PREK_;
    build_di<<<grid1(n), 256, 0, stream>>>(dec_in, di);
    const int PM = T_OUT_ * B_;  // 32000 rows
    wmma_gemm_bf16<<<dim3(PRE_ / 16 / 8, PM / 16), 256, 0, stream>>>(
        di, PREK_, p1_bf, PREK_, 0, nullptr, ptmp, PRE_, PM, PRE_, PREK_, 0);
    n = (long long)PM * PRE_;
    relu_drop_bf16<<<grid1(n), 256, 0, stream>>>(ptmp, x_bf, n, 0x9E3779B9u);
    wmma_gemm_bf16<<<dim3(PRE_ / 16 / 8, PM / 16), 256, 0, stream>>>(
        x_bf, PRE_, p2_bf, PRE_, 0, nullptr, ptmp, PRE_, PM, PRE_, PRE_, 0);
    relu_drop_bf16<<<grid1(n), 256, 0, stream>>>(ptmp, x_bf, n, 0x85EBCA77u);

    // zero recurrent state (fp32 + bf16 blocks)
    n = (long long)(4 * B_ * RNN_ + 3 * B_ * T_IN_);
    zero_u32<<<grid1(n), 256, 0, stream>>>((unsigned*)(ws + o_state), n);
    n = (long long)(2 * B_ * RNN_ + B_ * ENC_) / 2;   // bf16 pairs as u32
    zero_u32<<<grid1(n), 256, 0, stream>>>((unsigned*)(ws + o_sbf), n);

    // ---- 500 sequential decoder steps ----
    const dim3 gGrid(GATES_ / 16 / 8, B_ / 16);   // (32, 4) blocks of 8 waves over 64x4096 gates
    for (int t = 0; t < T_OUT_; ++t) {
        const bf16* x_t = x_bf + (size_t)t * B_ * PRE_;
        // attention LSTM gates = [x,ctx]@wih^T + bih + ah@whh^T   (bhh added in cell)
        wmma_gemm_bf16<<<gGrid, 256, 0, stream>>>(x_t,    PRE_, wih_a, 768,  0,    att_bih, gates, GATES_, B_, GATES_, PRE_, 0);
        wmma_gemm_bf16<<<gGrid, 256, 0, stream>>>(ctx_bf, ENC_, wih_a, 768,  PRE_, nullptr, gates, GATES_, B_, GATES_, ENC_, 1);
        wmma_gemm_bf16<<<gGrid, 256, 0, stream>>>(ah_bf,  RNN_, whh_a, RNN_, 0,    nullptr, gates, GATES_, B_, GATES_, RNN_, 1);
        lstm_cell<<<dim3(B_ * RNN_ / 256), 256, 0, stream>>>(gates, att_bhh, ah, ac, ah_bf);
        // location-sensitive attention
        attention_step<<<dim3(B_), 256, 0, stream>>>(
            ah, wq, loc_conv, loc_dense, vvec, pm, memory, mlen,
            aw, awc, ctx, ctx_bf,
            out + (size_t)B_ * T_OUT_ * N_MEL_ + (size_t)B_ * T_OUT_, t);
        // decoder LSTM gates = [ah,ctx]@wih^T + bih + dh@whh^T
        wmma_gemm_bf16<<<gGrid, 256, 0, stream>>>(ah_bf,  RNN_, wih_d, 1536, 0,    dec_bih, gates, GATES_, B_, GATES_, RNN_, 0);
        wmma_gemm_bf16<<<gGrid, 256, 0, stream>>>(ctx_bf, ENC_, wih_d, 1536, RNN_, nullptr, gates, GATES_, B_, GATES_, ENC_, 1);
        wmma_gemm_bf16<<<gGrid, 256, 0, stream>>>(dh_bf,  RNN_, whh_d, RNN_, 0,    nullptr, gates, GATES_, B_, GATES_, RNN_, 1);
        lstm_cell<<<dim3(B_ * RNN_ / 256), 256, 0, stream>>>(gates, dec_bhh, dh, dc, dh_bf);
        // mel + gate projection
        proj_step<<<dim3(B_), 128, 0, stream>>>(dh, ctx, proj_w, proj_b, gate_w, gate_b, out, t);
    }
}
